// Encoder_18313740550189
// MI455X (gfx1250) — matrix operand order repaired and checked by reading, not compiled, not run on hardware
//
#include <hip/hip_runtime.h>

// ---------------------------------------------------------------------------
// MI455X ConvLSTM encoder: implicit-GEMM via v_wmma_f32_16x16x32_bf16.
// ~503 GFLOP total vs ~60MB working set => compute bound; bf16 WMMA path.
// Round 3: register double-buffered K-loop so loads for chunk ci+1 are in
// flight while chunk ci's 8 WMMAs run (partial s_wait_loadcnt, not drains).
// ---------------------------------------------------------------------------

typedef __attribute__((ext_vector_type(16))) __bf16 v16bf;
typedef __attribute__((ext_vector_type(8)))  float  v8f;

#define NPIX   32768           // B*H*W = 8*64*64
#define HID    64
#define NGATE  256             // 4*HID
#define PH     66              // padded H
#define PW     66              // padded W
#define MTILE  64              // pixels per block (8 waves: 2 in M x 4 in N)

struct Bits32 { uint4 lo, hi; };   // 32 bytes -> one v16bf fragment

// ---------------------------------------------------------------------------
// Weight prep: HWIO f32 [3,3,CIN,256] -> bf16 [256][9*CPAD] (zero-pad channels)
// ---------------------------------------------------------------------------
template<int CIN, int CPAD>
__global__ void prep_weights(const float* __restrict__ W, __bf16* __restrict__ Wt) {
    int idx = blockIdx.x * blockDim.x + threadIdx.x;
    const int total = NGATE * 9 * CPAD;
    if (idx >= total) return;
    int n = idx / (9 * CPAD);
    int r = idx % (9 * CPAD);
    int t = r / CPAD;
    int c = r % CPAD;
    float v = (c < CIN) ? W[((size_t)(t * CIN + c)) * NGATE + n] : 0.0f;
    Wt[idx] = (__bf16)v;
}

// ---------------------------------------------------------------------------
// Per-step pack: x_t -> pack0[ch0:16]; h0(from pack1 ch0:64) -> pack0[ch16:80];
//                h1 (f32, from d_out) -> pack1[ch64:128]. Borders stay zero.
// ---------------------------------------------------------------------------
__global__ void pack_step(const float* __restrict__ enc, int t,
                          const __bf16* __restrict__ pack1_src,
                          const float* __restrict__ h1f32,
                          __bf16* __restrict__ pack0,
                          __bf16* __restrict__ pack1) {
    int idx = blockIdx.x * blockDim.x + threadIdx.x;
    const int total = NPIX * 144;                 // 16 + 64 + 64 channels/pixel
    if (idx >= total) return;
    int pix = idx / 144;
    int c   = idx % 144;
    int x = pix & 63, y = (pix >> 6) & 63, b = pix >> 12;
    size_t pp = ((size_t)(b * PH) + y + 1) * PW + x + 1;   // padded pixel
    if (c < 16) {
        size_t eidx = ((size_t)(b * 16 + t) * 4096 + (pix & 4095)) * 16 + c;
        pack0[pp * 96 + c] = (__bf16)enc[eidx];
    } else if (c < 80) {
        pack0[pp * 96 + c] = pack1_src[pp * 128 + (c - 16)];
    } else {
        int hid = c - 80;
        pack1[pp * 128 + 64 + hid] = (__bf16)h1f32[(size_t)pix * HID + hid];
    }
}

// ---------------------------------------------------------------------------
// Fused gate-conv (implicit GEMM, bf16 WMMA) + LSTM pointwise update.
// Block = 256 threads = 8 waves: 64 pixels x 256 gate channels per block.
// Wave (mw = wave/4, nq = wave%4): 32 pixels x 64 cols -> 8 16x16 accumulators;
// B fragments shared across both M halves; K-loop register double-buffered.
// ---------------------------------------------------------------------------
template<int CPAD>
__global__ void __launch_bounds__(256)
convlstm_gates(const __bf16* __restrict__ packIn,   // [B,66,66,CPAD]
               const __bf16* __restrict__ Wt,       // [256][9*CPAD]
               const float*  __restrict__ bias,     // [256]
               float*        __restrict__ cbuf,     // [NPIX,64] read+write
               __bf16*       __restrict__ hDestBf,  // pack1 (stride 128) or null
               int hOff,
               float*        __restrict__ hDestF32) // d_out h-half or null
{
    __shared__ float zsh[MTILE * NGATE];            // 64 KB gate staging

    const int lane  = threadIdx.x & 31;
    const int wave  = threadIdx.x >> 5;
    const int g     = lane >> 4;                    // lane group (K split)
    const int ncol  = lane & 15;
    const int nBase = (wave & 3) << 6;              // 0,64,128,192
    const int mBase = blockIdx.x * MTILE;           // 64 pixels / block
    const int mWave = ((wave >> 2) << 5);           // 0 or 32 within tile

    // bias folded into accumulator init (N = per-lane constant in C/D layout)
    v8f acc0[4], acc1[4];
    #pragma unroll
    for (int nn = 0; nn < 4; ++nn) {
        const float bv = bias[nBase + nn * 16 + ncol];
        v8f bvv = {bv, bv, bv, bv, bv, bv, bv, bv};
        acc0[nn] = bvv;
        acc1[nn] = bvv;
    }

    // ---- implicit GEMM over K = 9 taps * CPAD channels (double-buffered) ----
    {
        const int p0 = mBase + mWave + ncol;        // A rows for M-subtile 0
        const int p1 = p0 + 16;                     // A rows for M-subtile 1
        const int x0 = p0 & 63, y0 = (p0 >> 6) & 63, b0 = p0 >> 12;
        const int x1 = p1 & 63, y1 = (p1 >> 6) & 63, b1 = p1 >> 12;
        const __bf16* A0 = packIn + ((size_t)(b0 * PH + y0) * PW + x0) * CPAD;
        const __bf16* A1 = packIn + ((size_t)(b1 * PH + y1) * PW + x1) * CPAD;

        const __bf16* bptr[4];
        #pragma unroll
        for (int nn = 0; nn < 4; ++nn)
            bptr[nn] = Wt + (size_t)(nBase + nn * 16 + ncol) * (9 * CPAD) + g * 16;

        constexpr int CHT = CPAD / 32;              // chunks per tap
        constexpr int NCH = 9 * CHT;                // total K chunks

        auto loadA = [&](int ci, v16bf& a0, v16bf& a1) {
            const int tap = ci / CHT;
            const int kc  = (ci % CHT) * 32;
            const int off = (tap / 3) * PW + (tap % 3);   // ky*PW + kx
            const uint4* ap0 = (const uint4*)(A0 + off * CPAD + kc);
            const uint4* ap1 = (const uint4*)(A1 + off * CPAD + kc);
            Bits32 t0; t0.lo = ap0[2 * g]; t0.hi = ap0[2 * g + 1];
            Bits32 t1; t1.lo = ap1[2 * g]; t1.hi = ap1[2 * g + 1];
            a0 = __builtin_bit_cast(v16bf, t0);
            a1 = __builtin_bit_cast(v16bf, t1);
        };
        auto loadB = [&](int ci, v16bf bf[4]) {
            const int tap = ci / CHT;
            const int kc  = (ci % CHT) * 32;
            const int koff = tap * CPAD + kc;
            #pragma unroll
            for (int nn = 0; nn < 4; ++nn) {
                const uint4* bp = (const uint4*)(bptr[nn] + koff);
                Bits32 tb; tb.lo = bp[0]; tb.hi = bp[1];
                bf[nn] = __builtin_bit_cast(v16bf, tb);
            }
        };

        v16bf curA0, curA1, curB[4];
        v16bf nxtA0, nxtA1, nxtB[4];
        loadA(0, curA0, curA1);
        loadB(0, curB);

        #pragma unroll
        for (int ci = 0; ci < NCH; ++ci) {
            if (ci + 1 < NCH) {                     // prefetch next chunk
                loadA(ci + 1, nxtA0, nxtA1);
                loadB(ci + 1, nxtB);
            }
            #pragma unroll
            for (int nn = 0; nn < 4; ++nn) {
                acc0[nn] = __builtin_amdgcn_wmma_f32_16x16x32_bf16(
                    false, curA0, false, curB[nn], (short)0, acc0[nn], false, false);
                acc1[nn] = __builtin_amdgcn_wmma_f32_16x16x32_bf16(
                    false, curA1, false, curB[nn], (short)0, acc1[nn], false, false);
            }
            curA0 = nxtA0;
            curA1 = nxtA1;
            #pragma unroll
            for (int nn = 0; nn < 4; ++nn) curB[nn] = nxtB[nn];
        }
    }

    // ---- stage z into LDS; C/D layout: VGPR j, lane -> M=j+8*(lane/16), N=lane%16
    #pragma unroll
    for (int nn = 0; nn < 4; ++nn) {
        const int nch = nBase + nn * 16 + ncol;
        #pragma unroll
        for (int j = 0; j < 8; ++j) {
            const int m = mWave + (g << 3) + j;
            zsh[m * NGATE + nch]        = acc0[nn][j];
            zsh[(m + 16) * NGATE + nch] = acc1[nn][j];
        }
    }
    __syncthreads();

    // ---- fused LSTM pointwise update: 64 pixels * 64 hid = 4096 elems ----
    #pragma unroll
    for (int k = 0; k < 16; ++k) {
        const int flat = (k << 8) + threadIdx.x;
        const int pix = flat >> 6;
        const int hid = flat & 63;
        float iv = zsh[pix * NGATE +       hid];
        float fv = zsh[pix * NGATE +  64 + hid];
        float ov = zsh[pix * NGATE + 128 + hid];
        float gv = zsh[pix * NGATE + 192 + hid];
        iv = 1.0f / (1.0f + __expf(-iv));
        fv = 1.0f / (1.0f + __expf(-fv));
        ov = 1.0f / (1.0f + __expf(-ov));
        gv = tanhf(gv);
        const int gp = mBase + pix;
        const size_t cidx = ((size_t)gp << 6) + hid;
        const float cold = cbuf[cidx];
        const float cnew = fv * cold + iv * gv;
        const float hnew = ov * tanhf(cnew);
        cbuf[cidx] = cnew;
        if (hDestBf) {
            const int xx = gp & 63, yy = (gp >> 6) & 63, bb = gp >> 12;
            const size_t pp = ((size_t)(bb * PH) + yy + 1) * PW + xx + 1;
            hDestBf[pp * 128 + hOff + hid] = (__bf16)hnew;
        }
        if (hDestF32) hDestF32[cidx] = hnew;
    }
}

// ---------------------------------------------------------------------------
extern "C" void kernel_launch(void* const* d_in, const int* in_sizes, int n_in,
                              void* d_out, int out_size, void* d_ws, size_t ws_size,
                              hipStream_t stream) {
    (void)in_sizes; (void)n_in; (void)out_size; (void)ws_size;
    const float* enc = (const float*)d_in[0];
    const float* W0  = (const float*)d_in[1];
    const float* b0  = (const float*)d_in[2];
    const float* W1  = (const float*)d_in[3];
    const float* b1  = (const float*)d_in[4];
    float* out = (float*)d_out;

    // workspace carve-out (256B aligned)
    char* ws = (char*)d_ws;
    size_t off = 0;
    auto carve = [&](size_t bytes) {
        void* p = ws + off;
        off = (off + bytes + 255) & ~(size_t)255;
        return p;
    };
    const size_t pack0_bytes = (size_t)8 * PH * PW * 96  * sizeof(__bf16);
    const size_t pack1_bytes = (size_t)8 * PH * PW * 128 * sizeof(__bf16);
    const size_t c0_bytes    = (size_t)NPIX * HID * sizeof(float);
    __bf16* pack0 = (__bf16*)carve(pack0_bytes);
    __bf16* pack1 = (__bf16*)carve(pack1_bytes);
    float*  c0    = (float*) carve(c0_bytes);
    __bf16* W0t   = (__bf16*)carve((size_t)NGATE * 9 * 96  * sizeof(__bf16));
    __bf16* W1t   = (__bf16*)carve((size_t)NGATE * 9 * 128 * sizeof(__bf16));

    float* h1out = out;                         // [NPIX,64] f32
    float* c1    = out + (size_t)NPIX * HID;    // [NPIX,64] f32

    // deterministic re-init every launch (graph-capture safe memset nodes)
    hipMemsetAsync(pack0, 0, pack0_bytes, stream);
    hipMemsetAsync(pack1, 0, pack1_bytes, stream);
    hipMemsetAsync(c0,    0, c0_bytes,    stream);
    hipMemsetAsync(out,   0, (size_t)NPIX * 2 * HID * sizeof(float), stream);

    prep_weights<80, 96 ><<<(NGATE * 9 * 96  + 255) / 256, 256, 0, stream>>>(W0, W0t);
    prep_weights<128,128><<<(NGATE * 9 * 128 + 255) / 256, 256, 0, stream>>>(W1, W1t);

    const int packGrid = (NPIX * 144 + 255) / 256;
    const int convGrid = NPIX / MTILE;          // 512 blocks, 256 threads

    for (int t = 0; t < 16; ++t) {
        pack_step<<<packGrid, 256, 0, stream>>>(enc, t, pack1, h1out, pack0, pack1);
        // layer 0: reads pack0, h0 -> pack1[ch0:64]
        convlstm_gates<96 ><<<convGrid, 256, 0, stream>>>(
            pack0, W0t, b0, c0, pack1, 0, nullptr);
        // layer 1: reads pack1, h1/c1 -> d_out (f32)
        convlstm_gates<128><<<convGrid, 256, 0, stream>>>(
            pack1, W1t, b1, c1, nullptr, 0, h1out);
    }
}
